// GPT_26233660244182
// MI455X (gfx1250) — matrix operand order repaired and checked by reading, not compiled, not run on hardware
//
#include <hip/hip_runtime.h>
#include <hip/hip_bf16.h>

typedef __bf16 bf16_t;
typedef __attribute__((ext_vector_type(16))) __bf16 v16bf;
typedef __attribute__((ext_vector_type(8)))  __bf16 v8bf;
typedef __attribute__((ext_vector_type(8)))  float  v8f;

#define BD  2
#define TD  4096
#define DD  512
#define HH  8
#define HDD 64
#define KSEL 256
#define LL  4
#define VOC 32000
#define MD  (BD*TD)

#define LSTR 40                 // padded LDS row stride in halves (80 B)
#define LBUF (64 * LSTR * 2)    // one LDS tile buffer in bytes (5120)

// ---------------------------------------------------------------------------
__global__ void cvt_f32_bf16(const float* __restrict__ in, bf16_t* __restrict__ out, int n) {
    int i = blockIdx.x * 256 + threadIdx.x;
    if (i < n) out[i] = (bf16_t)in[i];
}

__global__ void fill_bf16_zero(bf16_t* __restrict__ p, int n) {
    int i = blockIdx.x * 256 + threadIdx.x;
    if (i < n) p[i] = (bf16_t)0.0f;
}

// x = E[tok] + P
__global__ void embed_kernel(const int* __restrict__ tok, const float* __restrict__ E,
                             const float* __restrict__ P, float* __restrict__ x) {
    int i = blockIdx.x * 256 + threadIdx.x;            // over M*D
    if (i >= MD * DD) return;
    int d  = i & (DD - 1);
    int bt = i / DD;
    int t  = bt & (TD - 1);
    x[i] = E[(size_t)tok[bt] * DD + d] + P[t * DD + d];
}

// RMSNorm row -> bf16
__global__ __launch_bounds__(256)
void rmsnorm_bf16(const float* __restrict__ x, const float* __restrict__ w,
                  bf16_t* __restrict__ out) {
    int row = blockIdx.x;
    const float* xr = x + (size_t)row * DD;
    __shared__ float red[256];
    float s = 0.f;
    for (int d = threadIdx.x; d < DD; d += 256) { float v = xr[d]; s += v * v; }
    red[threadIdx.x] = s; __syncthreads();
    for (int off = 128; off > 0; off >>= 1) {
        if (threadIdx.x < off) red[threadIdx.x] += red[threadIdx.x + off];
        __syncthreads();
    }
    float rs = rsqrtf(red[0] * (1.0f / DD) + 1.1920929e-7f);
    for (int d = threadIdx.x; d < DD; d += 256)
        out[(size_t)row * DD + d] = (bf16_t)(xr[d] * rs * w[d]);
}

// router scores r[b,h,t] = xn[b,t,:] . Wr[h,:]
__global__ void router_kernel(const bf16_t* __restrict__ xn, const float* __restrict__ Wr,
                              float* __restrict__ r) {
    int bt = blockIdx.x * 256 + threadIdx.x;
    if (bt >= MD) return;
    int b = bt / TD, t = bt & (TD - 1);
    float acc[HH] = {};
    const bf16_t* xr = xn + (size_t)bt * DD;
    for (int k = 0; k < DD; ++k) {
        float xv = (float)xr[k];
#pragma unroll
        for (int h = 0; h < HH; ++h) acc[h] += xv * Wr[h * DD + k];
    }
#pragma unroll
    for (int h = 0; h < HH; ++h) r[((size_t)(b * HH + h)) * TD + t] = acc[h];
}

// top-K per (b,h): iterative argmax (only the selected *set* matters:
// scatter + masked softmax are order-invariant for a fixed set)
__global__ __launch_bounds__(256)
void topk_kernel(const float* __restrict__ r, int* __restrict__ idx) {
    int bh = blockIdx.x;
    __shared__ float sv[TD];
    __shared__ float rv[256];
    __shared__ int   ri[256];
    for (int j = threadIdx.x; j < TD; j += 256) sv[j] = r[(size_t)bh * TD + j];
    __syncthreads();
    for (int it = 0; it < KSEL; ++it) {
        float best = -INFINITY; int bi = 0;
        for (int j = threadIdx.x; j < TD; j += 256)
            if (sv[j] > best) { best = sv[j]; bi = j; }
        rv[threadIdx.x] = best; ri[threadIdx.x] = bi; __syncthreads();
        for (int off = 128; off > 0; off >>= 1) {
            if (threadIdx.x < off && rv[threadIdx.x + off] > rv[threadIdx.x]) {
                rv[threadIdx.x] = rv[threadIdx.x + off];
                ri[threadIdx.x] = ri[threadIdx.x + off];
            }
            __syncthreads();
        }
        if (threadIdx.x == 0) { idx[bh * KSEL + it] = ri[0]; sv[ri[0]] = -INFINITY; }
        __syncthreads();
    }
}

// ---------------------------------------------------------------------------
// WMMA bf16 GEMM, block = 4 waves = 64Mx64N tile.
// Per 32-K stage: A(64x32) and B(64x32) bf16 tiles staged in LDS via
// ASYNCcnt-tracked global_load_async_to_lds_b128, double-buffered so the
// next stage's DMA overlaps this stage's ds_load + v_wmma work.
__global__ __launch_bounds__(128)
void gemm_bf16_wmma(const bf16_t* __restrict__ A, const bf16_t* __restrict__ W,
                    const float* __restrict__ bias, const float* __restrict__ resid,
                    float* __restrict__ outF, bf16_t* __restrict__ outB,
                    int M, int N, int K, int doSilu) {
    __shared__ __align__(16) bf16_t sA[2][64 * LSTR];
    __shared__ __align__(16) bf16_t sB[2][64 * LSTR];

    const int tid  = threadIdx.x;
    const int lane = tid & 31;
    const int wave = tid >> 5;
    const int l15  = lane & 15;
    const int tileM = blockIdx.y * 64;
    const int tileN = blockIdx.x * 64;

    // async-copy assignment: thread t copies rows (t>>2) and (t>>2)+32,
    // 16-byte column chunk (t&3) of both the A and B 64x32 tiles.
    const int crow = tid >> 2;
    const int ccol = (tid & 3) * 8;            // halves
    const bf16_t* gA0 = A + (size_t)(tileM + crow) * K + ccol;
    const bf16_t* gA1 = gA0 + (size_t)32 * K;
    const bf16_t* gB0 = W + (size_t)(tileN + crow) * K + ccol;
    const bf16_t* gB1 = gB0 + (size_t)32 * K;
    const unsigned ldsOff = (unsigned)(crow * LSTR + ccol) * 2;
    const unsigned dA = (unsigned)(uintptr_t)&sA[0][0] + ldsOff;
    const unsigned dB = (unsigned)(uintptr_t)&sB[0][0] + ldsOff;
    const unsigned rowHalf = (unsigned)(32 * LSTR) * 2;

    auto issue = [&](int buf, int k0) {
        unsigned bo = (unsigned)buf * LBUF;
        unsigned long long a0 = (unsigned long long)(uintptr_t)(gA0 + k0);
        unsigned long long a1 = (unsigned long long)(uintptr_t)(gA1 + k0);
        unsigned long long b0 = (unsigned long long)(uintptr_t)(gB0 + k0);
        unsigned long long b1 = (unsigned long long)(uintptr_t)(gB1 + k0);
        asm volatile("global_load_async_to_lds_b128 %0, %1, off"
                     :: "v"(dA + bo), "v"(a0) : "memory");
        asm volatile("global_load_async_to_lds_b128 %0, %1, off"
                     :: "v"(dA + bo + rowHalf), "v"(a1) : "memory");
        asm volatile("global_load_async_to_lds_b128 %0, %1, off"
                     :: "v"(dB + bo), "v"(b0) : "memory");
        asm volatile("global_load_async_to_lds_b128 %0, %1, off"
                     :: "v"(dB + bo + rowHalf), "v"(b1) : "memory");
    };

    v8f acc[4] = {};
    const int kbA = (lane >> 4) << 3;   // 0 / 8   (A: 16x32 bf16 fragment layout)
    const int kbB = (lane >> 4) << 3;   // 0 / 8   (B: 32x16 bf16 fragment layout)
    const int aRow = wave * 16 + l15;   // this wave's A row inside the LDS tile

    const int nstages = K >> 5;
    issue(0, 0);
    for (int s = 0; s < nstages; ++s) {
        int buf = s & 1;
        if (s + 1 < nstages) {
            issue(buf ^ 1, (s + 1) << 5);
            asm volatile("s_wait_asynccnt 4" ::: "memory");   // stage s landed
        } else {
            asm volatile("s_wait_asynccnt 0" ::: "memory");
        }
        __syncthreads();

        v16bf a;
        {
            const bf16_t* ap = &sA[buf][aRow * LSTR + kbA];
            v8bf lo = *(const v8bf*)(ap);
            v8bf hi = *(const v8bf*)(ap + 16);
#pragma unroll
            for (int i = 0; i < 8; ++i) { a[i] = lo[i]; a[i + 8] = hi[i]; }
        }
#pragma unroll
        for (int j = 0; j < 4; ++j) {
            const bf16_t* bp = &sB[buf][(16 * j + l15) * LSTR + kbB];
            v8bf blo = *(const v8bf*)(bp);
            v8bf bhi = *(const v8bf*)(bp + 16);
            v16bf b;
#pragma unroll
            for (int i = 0; i < 8; ++i) { b[i] = blo[i]; b[i + 8] = bhi[i]; }
            acc[j] = __builtin_amdgcn_wmma_f32_16x16x32_bf16(
                false, a, false, b, (short)0, acc[j], false, false);
        }
        __syncthreads();   // everyone done reading buf before it is re-filled
    }

    const int mbase = tileM + wave * 16 + ((lane >> 4) << 3);
#pragma unroll
    for (int j = 0; j < 4; ++j) {
        int n  = tileN + 16 * j + l15;
        float bv = bias ? bias[n] : 0.f;
#pragma unroll
        for (int rr = 0; rr < 8; ++rr) {
            float v = acc[j][rr] + bv;
            if (doSilu) v = v / (1.f + __expf(-v));
            size_t o = (size_t)(mbase + rr) * N + n;
            if (resid) v += resid[o];
            if (outF) outF[o] = v;
            if (outB) outB[o] = (bf16_t)v;
        }
    }
}

// ---------------------------------------------------------------------------
// sparse attention over selected indices; one block per (b,h); thread = query
__global__ __launch_bounds__(256)
void attn_kernel(const bf16_t* __restrict__ q, const bf16_t* __restrict__ k,
                 const bf16_t* __restrict__ v, const int* __restrict__ idx,
                 bf16_t* __restrict__ out) {
    int bh = blockIdx.x;
    int b  = bh / HH, h = bh % HH;
    int tid = threadIdx.x;

    __shared__ __align__(16) bf16_t kvS[2 * KSEL * HDD];   // 64 KB
    bf16_t* kS = kvS;
    bf16_t* vS = kvS + KSEL * HDD;

    // phase 1: selected-index overlay -> per-thread causal masks
    int* sidxO = (int*)kvS;
    sidxO[tid] = idx[bh * KSEL + tid];
    __syncthreads();
    int myti = sidxO[tid];
    unsigned mask[8];
#pragma unroll
    for (int w = 0; w < 8; ++w) {
        unsigned mw = 0;
        for (int bb = 0; bb < 32; ++bb)
            mw |= (myti >= sidxO[w * 32 + bb]) ? (1u << bb) : 0u;
        mask[w] = mw;
    }
    __syncthreads();

    // phase 2: gather K/V rows into LDS (128 B per row, uint4 copies)
    size_t base = ((size_t)(b * TD + myti)) * DD + h * HDD;
    {
        const uint4* ksrc = (const uint4*)(k + base);
        const uint4* vsrc = (const uint4*)(v + base);
        uint4* kdst = (uint4*)(kS + (size_t)tid * HDD);
        uint4* vdst = (uint4*)(vS + (size_t)tid * HDD);
#pragma unroll
        for (int w = 0; w < 8; ++w) { kdst[w] = ksrc[w]; vdst[w] = vsrc[w]; }
    }
    float qv[HDD];
    {
        const bf16_t* qs = q + base;
#pragma unroll
        for (int d = 0; d < HDD; ++d) qv[d] = (float)qs[d];
    }
    __syncthreads();

    // online softmax over selected keys
    float m = -INFINITY, l = 0.f, o[HDD] = {};
    for (int j = 0; j < KSEL; ++j) {
        if (!((mask[j >> 5] >> (j & 31)) & 1u)) continue;
        const bf16_t* kr = kS + j * HDD;
        float s = 0.f;
#pragma unroll
        for (int d = 0; d < HDD; ++d) s += qv[d] * (float)kr[d];
        s *= 0.125f;
        float p;
        if (s > m) {
            float c = __expf(m - s);
            l *= c;
#pragma unroll
            for (int d = 0; d < HDD; ++d) o[d] *= c;
            m = s; p = 1.f;
        } else {
            p = __expf(s - m);
        }
        l += p;
        const bf16_t* vr = vS + j * HDD;
#pragma unroll
        for (int d = 0; d < HDD; ++d) o[d] += p * (float)vr[d];
    }
    float inv = 1.f / l;
    bf16_t* od = out + base;
#pragma unroll
    for (int d = 0; d < HDD; ++d) od[d] = (bf16_t)(o[d] * inv);
}

// ---------------------------------------------------------------------------
static inline void run_gemm(const bf16_t* A, const bf16_t* W, const float* bias,
                            const float* resid, float* outF, bf16_t* outB,
                            int M, int N, int K, int silu, hipStream_t s) {
    dim3 g(N / 64, M / 64), b(128);
    gemm_bf16_wmma<<<g, b, 0, s>>>(A, W, bias, resid, outF, outB, M, N, K, silu);
}

extern "C" void kernel_launch(void* const* d_in, const int* in_sizes, int n_in,
                              void* d_out, int out_size, void* d_ws, size_t ws_size,
                              hipStream_t stream) {
    (void)in_sizes; (void)n_in; (void)out_size; (void)ws_size;
    const int*   tok = (const int*)  d_in[0];
    const float* E   = (const float*)d_in[1];
    const float* P   = (const float*)d_in[2];
    const float* Wr  = (const float*)d_in[3];
    const float* Wq  = (const float*)d_in[4];
    const float* Wk  = (const float*)d_in[5];
    const float* Wv  = (const float*)d_in[6];
    const float* Wo  = (const float*)d_in[7];
    const float* n1  = (const float*)d_in[8];
    const float* n2  = (const float*)d_in[9];
    const float* W1  = (const float*)d_in[10];
    const float* b1  = (const float*)d_in[11];
    const float* W2  = (const float*)d_in[12];
    const float* b2  = (const float*)d_in[13];
    float* logits = (float*)d_out;

    const int D4 = 4 * DD;
    char* ws = (char*)d_ws; size_t off = 0;
    auto alloc = [&](size_t bytes) -> void* {
        void* p = ws + off; off = (off + bytes + 255) & ~(size_t)255; return p;
    };
    bf16_t* Ebf  = (bf16_t*)alloc((size_t)VOC * DD * 2);
    bf16_t* Wqb  = (bf16_t*)alloc((size_t)LL * DD * DD * 2);
    bf16_t* Wkb  = (bf16_t*)alloc((size_t)LL * DD * DD * 2);
    bf16_t* Wvb  = (bf16_t*)alloc((size_t)LL * DD * DD * 2);
    bf16_t* Wob  = (bf16_t*)alloc((size_t)LL * DD * DD * 2);
    bf16_t* W1b  = (bf16_t*)alloc((size_t)LL * D4 * DD * 2);
    bf16_t* W2b  = (bf16_t*)alloc((size_t)LL * DD * D4 * 2);
    float*  x    = (float*) alloc((size_t)MD * DD * 4);
    bf16_t* xnb  = (bf16_t*)alloc((size_t)MD * DD * 2);
    bf16_t* qb   = (bf16_t*)alloc((size_t)MD * DD * 2);
    bf16_t* kb   = (bf16_t*)alloc((size_t)MD * DD * 2);
    bf16_t* vb   = (bf16_t*)alloc((size_t)MD * DD * 2);
    bf16_t* atb  = (bf16_t*)alloc((size_t)MD * DD * 2);
    bf16_t* h1b  = (bf16_t*)alloc((size_t)MD * D4 * 2);
    float*  rsc  = (float*) alloc((size_t)BD * HH * TD * 4);
    int*    idxb = (int*)   alloc((size_t)BD * HH * KSEL * 4);

    auto cvt = [&](const float* src, bf16_t* dst, int n) {
        cvt_f32_bf16<<<(n + 255) / 256, 256, 0, stream>>>(src, dst, n);
    };
    cvt(E,  Ebf, VOC * DD);
    cvt(Wq, Wqb, LL * DD * DD);
    cvt(Wk, Wkb, LL * DD * DD);
    cvt(Wv, Wvb, LL * DD * DD);
    cvt(Wo, Wob, LL * DD * DD);
    cvt(W1, W1b, LL * D4 * DD);
    cvt(W2, W2b, LL * DD * D4);

    embed_kernel<<<(MD * DD + 255) / 256, 256, 0, stream>>>(tok, E, P, x);

    for (int l = 0; l < LL; ++l) {
        rmsnorm_bf16<<<MD, 256, 0, stream>>>(x, n1 + l * DD, xnb);
        router_kernel<<<MD / 256, 256, 0, stream>>>(xnb, Wr + l * HH * DD, rsc);
        topk_kernel<<<BD * HH, 256, 0, stream>>>(rsc, idxb);

        run_gemm(xnb, Wqb + (size_t)l * DD * DD, nullptr, nullptr, nullptr, qb, MD, DD, DD, 0, stream);
        run_gemm(xnb, Wkb + (size_t)l * DD * DD, nullptr, nullptr, nullptr, kb, MD, DD, DD, 0, stream);
        run_gemm(xnb, Wvb + (size_t)l * DD * DD, nullptr, nullptr, nullptr, vb, MD, DD, DD, 0, stream);

        fill_bf16_zero<<<(MD * DD + 255) / 256, 256, 0, stream>>>(atb, MD * DD);
        attn_kernel<<<BD * HH, 256, 0, stream>>>(qb, kb, vb, idxb, atb);

        // x = x + attn @ Wo^T  (in-place residual, per-element read-then-write)
        run_gemm(atb, Wob + (size_t)l * DD * DD, nullptr, x, x, nullptr, MD, DD, DD, 0, stream);

        rmsnorm_bf16<<<MD, 256, 0, stream>>>(x, n2 + l * DD, xnb);
        // h1 = silu(xn @ W1^T + b1)
        run_gemm(xnb, W1b + (size_t)l * D4 * DD, b1 + l * D4, nullptr, nullptr, h1b, MD, D4, DD, 1, stream);
        // x = x + h1 @ W2^T + b2
        run_gemm(h1b, W2b + (size_t)l * DD * D4, b2 + l * DD, x, x, nullptr, MD, DD, D4, 0, stream);
    }

    // lm_head: logits = x @ E^T
    cvt(x, xnb, MD * DD);
    run_gemm(xnb, Ebf, nullptr, nullptr, logits, nullptr, MD, VOC, DD, 0, stream);
}
